// SimplifiedDAN_78391743087022
// MI455X (gfx1250) — compile-verified, matrix operand order checked by reading, not run on hardware
//
#include <hip/hip_runtime.h>
#include <hip/hip_bf16.h>
#include <math.h>

typedef __attribute__((ext_vector_type(16))) _Float16 v16h;
typedef __attribute__((ext_vector_type(8)))  _Float16 v8h;
typedef __attribute__((ext_vector_type(8)))  float    v8f;
typedef __attribute__((ext_vector_type(4)))  float    f32x4;

#define BATCH 4
#define SEQ   2048
#define DIM   1024
#define TBD   128
#define INV_SCALE 0.03125f   // 1/sqrt(1024)

#define QKV_ELEMS  ((size_t)BATCH * SEQ * DIM)   // 8,388,608
#define ATTN_ELEMS ((size_t)BATCH * SEQ * SEQ)   // 16,777,216

// ---- fragment loaders (CDNA5 16-bit A/B 16x16x32 layout) -------------------
// elems 0..7 = p[0..7], elems 8..15 = p[16..23]; caller applies lane ko.
__device__ __forceinline__ v16h load_frag_f16(const _Float16* __restrict__ p) {
    v8h lo = *(const v8h*)p;          // 16B -> global/ds b128
    v8h hi = *(const v8h*)(p + 16);
    v16h r;
#pragma unroll
    for (int i = 0; i < 8; ++i) { r[i] = lo[i]; r[8 + i] = hi[i]; }
    return r;
}
__device__ __forceinline__ v16h load_frag_f32(const float* __restrict__ p) {
    v16h r;
#pragma unroll
    for (int i = 0; i < 8; ++i) r[i] = (_Float16)p[i];
#pragma unroll
    for (int i = 0; i < 8; ++i) r[8 + i] = (_Float16)p[16 + i];
    return r;
}

#define WMMA16(A, B, C) \
    __builtin_amdgcn_wmma_f32_16x16x32_f16(false, A, false, B, (short)0, C, false, false)

// ---------------------------------------------------------------------------
// f32 -> f16 elementwise convert (8 elems/thread, b128 in / b128 out)
// ---------------------------------------------------------------------------
__global__ __launch_bounds__(256)
void dan_cvt_kernel(const float* __restrict__ src, _Float16* __restrict__ dst) {
    const size_t i = ((size_t)blockIdx.x * 256 + threadIdx.x) * 8;
    f32x4 x = *(const f32x4*)(src + i);
    f32x4 y = *(const f32x4*)(src + i + 4);
    v8h o;
#pragma unroll
    for (int j = 0; j < 4; ++j) { o[j] = (_Float16)x[j]; o[4 + j] = (_Float16)y[j]; }
    *(v8h*)(dst + i) = o;
}

// ---------------------------------------------------------------------------
// Scores (FAST path): raw Q@K^T from f16 copies. 64x64 block tile, 4 waves,
// 2x k-unroll with two independent fragment sets (software pipeline).
// ---------------------------------------------------------------------------
__global__ __launch_bounds__(128)
void dan_scores_f16_kernel(const _Float16* __restrict__ Qh,
                           const _Float16* __restrict__ Kh,
                           float* __restrict__ attn) {
    const int b     = blockIdx.z;
    const int qBase = blockIdx.y * 64;
    const int kBase = blockIdx.x * 64;
    const int wave  = threadIdx.x >> 5;
    const int lane  = threadIdx.x & 31;
    const int m     = lane & 15;
    const int ko    = (lane >> 4) * 8;

    const _Float16* __restrict__ Qrow = Qh + ((size_t)b * SEQ + qBase + wave * 16 + m) * DIM + ko;
    const _Float16* __restrict__ Kb   = Kh + ((size_t)b * SEQ + kBase + m) * DIM + ko;

    v8f acc0 = {}, acc1 = {}, acc2 = {}, acc3 = {};

    // prologue: fragment set X at d0 = 0
    v16h aX  = load_frag_f16(Qrow);
    v16h bX0 = load_frag_f16(Kb + (size_t)0  * DIM);
    v16h bX1 = load_frag_f16(Kb + (size_t)16 * DIM);
    v16h bX2 = load_frag_f16(Kb + (size_t)32 * DIM);
    v16h bX3 = load_frag_f16(Kb + (size_t)48 * DIM);

    for (int d0 = 0; d0 < DIM; d0 += 64) {
        // load set Y at d0+32 (always in range: DIM % 64 == 0)
        v16h aY  = load_frag_f16(Qrow + d0 + 32);
        v16h bY0 = load_frag_f16(Kb + (size_t)0  * DIM + d0 + 32);
        v16h bY1 = load_frag_f16(Kb + (size_t)16 * DIM + d0 + 32);
        v16h bY2 = load_frag_f16(Kb + (size_t)32 * DIM + d0 + 32);
        v16h bY3 = load_frag_f16(Kb + (size_t)48 * DIM + d0 + 32);

        acc0 = WMMA16(aX, bX0, acc0);
        acc1 = WMMA16(aX, bX1, acc1);
        acc2 = WMMA16(aX, bX2, acc2);
        acc3 = WMMA16(aX, bX3, acc3);

        if (d0 + 64 < DIM) {            // load set X at d0+64
            aX  = load_frag_f16(Qrow + d0 + 64);
            bX0 = load_frag_f16(Kb + (size_t)0  * DIM + d0 + 64);
            bX1 = load_frag_f16(Kb + (size_t)16 * DIM + d0 + 64);
            bX2 = load_frag_f16(Kb + (size_t)32 * DIM + d0 + 64);
            bX3 = load_frag_f16(Kb + (size_t)48 * DIM + d0 + 64);
        }
        acc0 = WMMA16(aY, bY0, acc0);
        acc1 = WMMA16(aY, bY1, acc1);
        acc2 = WMMA16(aY, bY2, acc2);
        acc3 = WMMA16(aY, bY3, acc3);
    }

    float* __restrict__ outBase = attn + ((size_t)b * SEQ + qBase + wave * 16) * SEQ + kBase;
    const int rOff = (lane >> 4) * 8;
    const int c    = lane & 15;
#pragma unroll
    for (int r = 0; r < 8; ++r) {
        float* row = outBase + (size_t)(r + rOff) * SEQ + c;
        row[0] = acc0[r]; row[16] = acc1[r]; row[32] = acc2[r]; row[48] = acc3[r];
    }
}

// ---------------------------------------------------------------------------
// Scores (FALLBACK path): same as round 1, f32 loads + in-register cvt.
// ---------------------------------------------------------------------------
__global__ __launch_bounds__(128)
void dan_scores_f32_kernel(const float* __restrict__ Q, const float* __restrict__ K,
                           float* __restrict__ attn) {
    const int b     = blockIdx.z;
    const int qBase = blockIdx.y * 64;
    const int kBase = blockIdx.x * 64;
    const int wave  = threadIdx.x >> 5;
    const int lane  = threadIdx.x & 31;
    const int m     = lane & 15;
    const int ko    = (lane >> 4) * 8;

    const float* __restrict__ Qrow = Q + ((size_t)b * SEQ + qBase + wave * 16 + m) * DIM + ko;
    const float* __restrict__ Kb   = K + ((size_t)b * SEQ + kBase + m) * DIM + ko;

    v8f acc0 = {}, acc1 = {}, acc2 = {}, acc3 = {};
    for (int d0 = 0; d0 < DIM; d0 += 32) {
        v16h a  = load_frag_f32(Qrow + d0);
        v16h b0 = load_frag_f32(Kb + (size_t)0  * DIM + d0);
        v16h b1 = load_frag_f32(Kb + (size_t)16 * DIM + d0);
        v16h b2 = load_frag_f32(Kb + (size_t)32 * DIM + d0);
        v16h b3 = load_frag_f32(Kb + (size_t)48 * DIM + d0);
        acc0 = WMMA16(a, b0, acc0);
        acc1 = WMMA16(a, b1, acc1);
        acc2 = WMMA16(a, b2, acc2);
        acc3 = WMMA16(a, b3, acc3);
    }
    float* __restrict__ outBase = attn + ((size_t)b * SEQ + qBase + wave * 16) * SEQ + kBase;
    const int rOff = (lane >> 4) * 8;
    const int c    = lane & 15;
#pragma unroll
    for (int r = 0; r < 8; ++r) {
        float* row = outBase + (size_t)(r + rOff) * SEQ + c;
        row[0] = acc0[r]; row[16] = acc1[r]; row[32] = acc2[r]; row[48] = acc3[r];
    }
}

// ---------------------------------------------------------------------------
// Gated row softmax in place; optionally also emits f16 weights for stage 3.
// ---------------------------------------------------------------------------
__global__ __launch_bounds__(256)
void dan_softmax_kernel(const float* __restrict__ tb, const float* __restrict__ W,
                        const float* __restrict__ bias, float* __restrict__ attn,
                        _Float16* __restrict__ attnH) {
    const int row = blockIdx.x;
    const int b   = row >> 11;
    const int t   = threadIdx.x;

    __shared__ float red[256];
    __shared__ float sh_scalar;

    float g = 0.0f;
    if (t < TBD) g = tb[b * TBD + t] * W[t];
    red[t] = g;
    __syncthreads();
#pragma unroll
    for (int off = 128; off > 0; off >>= 1) {
        if (t < off) red[t] += red[t + off];
        __syncthreads();
    }
    if (t == 0) {
        float z = red[0] + bias[0];
        sh_scalar = (1.0f / (1.0f + expf(-z))) * INV_SCALE;   // gate/sqrt(D)
    }
    __syncthreads();
    const float tscale = sh_scalar;

    float* __restrict__ prow = attn + (size_t)row * SEQ;
    float v[8];
    float mx = -INFINITY;
#pragma unroll
    for (int i = 0; i < 8; ++i) { v[i] = prow[t + 256 * i]; mx = fmaxf(mx, v[i]); }
    red[t] = mx;
    __syncthreads();
#pragma unroll
    for (int off = 128; off > 0; off >>= 1) {
        if (t < off) red[t] = fmaxf(red[t], red[t + off]);
        __syncthreads();
    }
    mx = red[0];
    __syncthreads();

    float s = 0.0f;
#pragma unroll
    for (int i = 0; i < 8; ++i) { v[i] = expf(tscale * (v[i] - mx)); s += v[i]; }
    red[t] = s;
    __syncthreads();
#pragma unroll
    for (int off = 128; off > 0; off >>= 1) {
        if (t < off) red[t] += red[t + off];
        __syncthreads();
    }
    const float inv = 1.0f / red[0];
    if (attnH) {
        _Float16* __restrict__ hrow = attnH + (size_t)row * SEQ;
#pragma unroll
        for (int i = 0; i < 8; ++i) {
            const float w = v[i] * inv;
            prow[t + 256 * i] = w;
            hrow[t + 256 * i] = (_Float16)w;
        }
    } else {
#pragma unroll
        for (int i = 0; i < 8; ++i) prow[t + 256 * i] = v[i] * inv;
    }
}

// ---------------------------------------------------------------------------
// Output (FAST path): attnH(f16) @ Vh(f16). V tile staged transposed in LDS.
// ---------------------------------------------------------------------------
__global__ __launch_bounds__(128)
void dan_out_f16_kernel(const _Float16* __restrict__ attnH,
                        const _Float16* __restrict__ Vh,
                        float* __restrict__ out) {
    const int b     = blockIdx.z;
    const int qBase = blockIdx.y * 64;
    const int dBase = blockIdx.x * 64;
    const int wave  = threadIdx.x >> 5;
    const int lane  = threadIdx.x & 31;
    const int m     = lane & 15;
    const int ko    = (lane >> 4) * 8;

    __shared__ _Float16 ldsB[64 * 32];   // [n][k]

    const _Float16* __restrict__ Arow = attnH + ((size_t)b * SEQ + qBase + wave * 16 + m) * SEQ + ko;
    const _Float16* __restrict__ Vb   = Vh + (size_t)b * SEQ * DIM;

    v8f acc0 = {}, acc1 = {}, acc2 = {}, acc3 = {};

    for (int k0 = 0; k0 < SEQ; k0 += 32) {
        __syncthreads();
#pragma unroll
        for (int i = threadIdx.x; i < 64 * 32; i += 128) {
            const int kk = i >> 6;
            const int n  = i & 63;
            ldsB[n * 32 + kk] = Vb[(size_t)(k0 + kk) * DIM + dBase + n];
        }
        __syncthreads();

        v16h a  = load_frag_f16(Arow + k0);
        v16h b0 = load_frag_f16(&ldsB[(0  + m) * 32 + ko]);
        v16h b1 = load_frag_f16(&ldsB[(16 + m) * 32 + ko]);
        v16h b2 = load_frag_f16(&ldsB[(32 + m) * 32 + ko]);
        v16h b3 = load_frag_f16(&ldsB[(48 + m) * 32 + ko]);
        acc0 = WMMA16(a, b0, acc0);
        acc1 = WMMA16(a, b1, acc1);
        acc2 = WMMA16(a, b2, acc2);
        acc3 = WMMA16(a, b3, acc3);
    }

    float* __restrict__ outBase = out + ((size_t)b * SEQ + qBase + wave * 16) * DIM + dBase;
    const int rOff = (lane >> 4) * 8;
    const int c    = lane & 15;
#pragma unroll
    for (int r = 0; r < 8; ++r) {
        float* row = outBase + (size_t)(r + rOff) * DIM + c;
        row[0] = acc0[r]; row[16] = acc1[r]; row[32] = acc2[r]; row[48] = acc3[r];
    }
}

// ---------------------------------------------------------------------------
// Output (FALLBACK path): f32 attn/V with in-register conversion via LDS.
// ---------------------------------------------------------------------------
__global__ __launch_bounds__(128)
void dan_out_f32_kernel(const float* __restrict__ attn, const float* __restrict__ V,
                        float* __restrict__ out) {
    const int b     = blockIdx.z;
    const int qBase = blockIdx.y * 64;
    const int dBase = blockIdx.x * 64;
    const int wave  = threadIdx.x >> 5;
    const int lane  = threadIdx.x & 31;
    const int m     = lane & 15;
    const int ko    = (lane >> 4) * 8;

    __shared__ _Float16 ldsB[64 * 32];

    const float* __restrict__ Arow = attn + ((size_t)b * SEQ + qBase + wave * 16 + m) * SEQ + ko;
    const float* __restrict__ Vb   = V + (size_t)b * SEQ * DIM;

    v8f acc0 = {}, acc1 = {}, acc2 = {}, acc3 = {};
    for (int k0 = 0; k0 < SEQ; k0 += 32) {
        __syncthreads();
#pragma unroll
        for (int i = threadIdx.x; i < 64 * 32; i += 128) {
            const int kk = i >> 6;
            const int n  = i & 63;
            ldsB[n * 32 + kk] = (_Float16)Vb[(size_t)(k0 + kk) * DIM + dBase + n];
        }
        __syncthreads();

        v16h a  = load_frag_f32(Arow + k0);
        v16h b0 = load_frag_f16(&ldsB[(0  + m) * 32 + ko]);
        v16h b1 = load_frag_f16(&ldsB[(16 + m) * 32 + ko]);
        v16h b2 = load_frag_f16(&ldsB[(32 + m) * 32 + ko]);
        v16h b3 = load_frag_f16(&ldsB[(48 + m) * 32 + ko]);
        acc0 = WMMA16(a, b0, acc0);
        acc1 = WMMA16(a, b1, acc1);
        acc2 = WMMA16(a, b2, acc2);
        acc3 = WMMA16(a, b3, acc3);
    }

    float* __restrict__ outBase = out + ((size_t)b * SEQ + qBase + wave * 16) * DIM + dBase;
    const int rOff = (lane >> 4) * 8;
    const int c    = lane & 15;
#pragma unroll
    for (int r = 0; r < 8; ++r) {
        float* row = outBase + (size_t)(r + rOff) * DIM + c;
        row[0] = acc0[r]; row[16] = acc1[r]; row[32] = acc2[r]; row[48] = acc3[r];
    }
}

extern "C" void kernel_launch(void* const* d_in, const int* in_sizes, int n_in,
                              void* d_out, int out_size, void* d_ws, size_t ws_size,
                              hipStream_t stream) {
    (void)in_sizes; (void)n_in; (void)out_size;

    const float* Q    = (const float*)d_in[0];
    const float* K    = (const float*)d_in[1];
    const float* V    = (const float*)d_in[2];
    const float* tb   = (const float*)d_in[3];
    const float* W    = (const float*)d_in[4];
    const float* bias = (const float*)d_in[5];

    float* out  = (float*)d_out;                               // [B,S,D]
    float* attn = (float*)d_out + QKV_ELEMS;                   // [B,S,S]

    const size_t need = (3 * QKV_ELEMS + ATTN_ELEMS) * sizeof(_Float16);  // 80 MB
    const bool fast = (d_ws != nullptr) && (ws_size >= need);

    dim3 gScores(SEQ / 64, SEQ / 64, BATCH);
    dim3 gOut(DIM / 64, SEQ / 64, BATCH);

    if (fast) {
        _Float16* Qh = (_Float16*)d_ws;
        _Float16* Kh = Qh + QKV_ELEMS;
        _Float16* Vh = Kh + QKV_ELEMS;
        _Float16* Ah = Vh + QKV_ELEMS;

        const int cvtBlocks = (int)(QKV_ELEMS / (256 * 8));    // 4096
        dan_cvt_kernel<<<cvtBlocks, 256, 0, stream>>>(Q, Qh);
        dan_cvt_kernel<<<cvtBlocks, 256, 0, stream>>>(K, Kh);
        dan_cvt_kernel<<<cvtBlocks, 256, 0, stream>>>(V, Vh);

        dan_scores_f16_kernel<<<gScores, 128, 0, stream>>>(Qh, Kh, attn);
        dan_softmax_kernel<<<BATCH * SEQ, 256, 0, stream>>>(tb, W, bias, attn, Ah);
        dan_out_f16_kernel<<<gOut, 128, 0, stream>>>(Ah, Vh, out);
    } else {
        dan_scores_f32_kernel<<<gScores, 128, 0, stream>>>(Q, K, attn);
        dan_softmax_kernel<<<BATCH * SEQ, 256, 0, stream>>>(tb, W, bias, attn, nullptr);
        dan_out_f32_kernel<<<gOut, 128, 0, stream>>>(attn, V, out);
    }
}
